// DecoderLSTM_678604833124
// MI455X (gfx1250) — matrix operand order repaired and checked by reading, not compiled, not run on hardware
//
#include <hip/hip_runtime.h>
#include <hip/hip_bf16.h>

// ---------------------------------------------------------------------------
// DecoderLSTM (Show-Attend-Tell) for MI455X / gfx1250, wave32, bf16 WMMA.
// B=64, L=196, CV=2048, T=20 (19 steps), VOCAB=10000, E=H=A=512.
// GEMMs: X[M][K] @ W[N][K]^T -> fp32 via V_WMMA_F32_16X16X32_BF16.
// W tiles are staged to LDS with GLOBAL_LOAD_ASYNC_TO_LDS (double-buffered),
// shared by 4 waves; each wave computes 1 M-tile x 4 N-tiles (A-frag reuse).
// V (103 MB fp32) stays L2-resident (192 MB L2) across the 19 steps.
// ---------------------------------------------------------------------------

#define BB    64
#define LL    196
#define CVD   2048
#define TT    20
#define NSTEP (TT - 1)      // 19
#define VOC   10000
#define ED    512
#define HD    512
#define AD    512
#define XK    (ED + CVD + HD)   // 3072: [emb | ctx | h]
#define G4H   (4 * HD)          // 2048

typedef __attribute__((ext_vector_type(8)))  float   v8f;
typedef __attribute__((ext_vector_type(16))) __bf16  bf16x16;
typedef __attribute__((ext_vector_type(8)))  __bf16  bf16x8;

// exact pointer types for the async-to-LDS builtin (param 1 is AS1 int4*)
typedef int v4i __attribute__((vector_size(16)));
typedef __attribute__((address_space(1))) v4i* as1_v4i_p;
typedef __attribute__((address_space(3))) v4i* as3_v4i_p;

#if __has_builtin(__builtin_amdgcn_global_load_async_to_lds_b128)
#define HAVE_ASYNC_LDS 1
#else
#define HAVE_ASYNC_LDS 0
#endif

#define SB_STRIDE 40   // bf16 units per LDS row (80 B; bank-conflict padding)

// Stage 64 rows x 32 K-values of W (bf16) into LDS. 128 threads: each thread
// copies 32 B (one row-half) as two B128 chunks. Rows clamped for the ragged
// last logits block (reads stay in-bounds; stores are guarded separately).
__device__ __forceinline__ void stage_B(const __bf16* __restrict__ W, int ldw,
                                        int Nrows, int n0, int kk,
                                        __bf16* sbuf, int tid)
{
    const int row  = tid >> 1;     // 0..63
    const int half = tid & 1;      // 0..1 (16 bf16 each)
    int r = n0 + row; if (r >= Nrows) r = Nrows - 1;
    const __bf16* g = W + (size_t)r * ldw + kk + half * 16;
    __bf16* l = sbuf + row * SB_STRIDE + half * 16;
#if HAVE_ASYNC_LDS
    __builtin_amdgcn_global_load_async_to_lds_b128(
        (as1_v4i_p)(void*)g, (as3_v4i_p)(void*)l, 0, 0);
    __builtin_amdgcn_global_load_async_to_lds_b128(
        (as1_v4i_p)(void*)(g + 8), (as3_v4i_p)(void*)(l + 8), 0, 0);
#else
    *(bf16x8*)l       = *(const bf16x8*)g;
    *(bf16x8*)(l + 8) = *(const bf16x8*)(g + 8);
#endif
}

__device__ __forceinline__ void wait_stage_async()
{
#if HAVE_ASYNC_LDS
#if __has_builtin(__builtin_amdgcn_s_wait_asynccnt)
    __builtin_amdgcn_s_wait_asynccnt(0);
#else
    asm volatile("s_wait_asynccnt 0x0" ::: "memory");
#endif
#endif
}

// ---------------------------------------------------------------------------
// WMMA GEMM: D[M][N] = A[M][K] * W[N][K]^T + bias[N]   (fp32 out)
// block = 128 (4 waves); wave w -> M-tile (blockIdx.x*4+w), N-tiles
// blockIdx.y*4 .. +3 (shared via LDS). A bf16, or fp32 converted on the fly.
// Fragment layouts (CDNA5 ISA 7.12.2):
//   A: lane m=(lane&15) = row M; K = {kb..kb+7, 16+kb..16+kb+7}, kb=(lane>>4)*8
//   B: lane n=(lane&15) = col N; K = {kb..kb+7, 16+kb..16+kb+7}, kb=(lane>>4)*8
//   C/D: lane n=(lane&15) col; VGPR r -> row M = (lane>>4)*8 + r
// ---------------------------------------------------------------------------
template <bool A_F32>
__global__ __launch_bounds__(128)
void wmma_gemm_bf16(const void* __restrict__ Aptr, int lda,
                    const __bf16* __restrict__ W, int ldw,
                    const float* __restrict__ bias,
                    float* __restrict__ D, long long ldd,
                    int Ntiles, int K)
{
    __shared__ __bf16 sB[2][64 * SB_STRIDE];

    const int tid    = threadIdx.x;
    const int lane   = tid & 31;
    const int wave   = tid >> 5;
    const int tile_m = blockIdx.x * 4 + wave;
    const int n0     = blockIdx.y * 64;     // first staged W row
    const int Nrows  = Ntiles * 16;

    const int mrow = tile_m * 16 + (lane & 15);
    const int akb  = (lane >> 4) * 8;

    const float*  Af = (const float*)Aptr;
    const __bf16* Ab = (const __bf16*)Aptr;

    v8f acc[4] = {};

    stage_B(W, ldw, Nrows, n0, 0, sB[0], tid);

    for (int kk = 0, it = 0; kk < K; kk += 32, ++it) {
        const int buf = it & 1;
        wait_stage_async();
        __syncthreads();
        if (kk + 32 < K) stage_B(W, ldw, Nrows, n0, kk + 32, sB[buf ^ 1], tid);

        // ---- A fragment (direct global) ----
        bf16x16 afrag;
        if (A_F32) {
            const float* ap = Af + (size_t)mrow * lda + kk + akb;
            const float4 f0 = *(const float4*)(ap);
            const float4 f1 = *(const float4*)(ap + 4);
            const float4 f2 = *(const float4*)(ap + 16);
            const float4 f3 = *(const float4*)(ap + 20);
            float fa[16] = { f0.x, f0.y, f0.z, f0.w,  f1.x, f1.y, f1.z, f1.w,
                             f2.x, f2.y, f2.z, f2.w,  f3.x, f3.y, f3.z, f3.w };
#pragma unroll
            for (int i = 0; i < 16; ++i) afrag[i] = (__bf16)fa[i];
        } else {
            const __bf16* ap = Ab + (size_t)mrow * lda + kk + akb;
            const bf16x8 lo = *(const bf16x8*)(ap);
            const bf16x8 hi = *(const bf16x8*)(ap + 16);
#pragma unroll
            for (int i = 0; i < 8; ++i) { afrag[i] = lo[i]; afrag[i + 8] = hi[i]; }
        }

        // ---- load all 4 B fragments first (one ds clause), then 4 WMMA ----
        const __bf16* sb = sB[buf];
        bf16x16 bfr[4];
#pragma unroll
        for (int j = 0; j < 4; ++j) {
            const __bf16* bp = sb + (j * 16 + (lane & 15)) * SB_STRIDE + (lane >> 4) * 8;
            const bf16x8 lo = *(const bf16x8*)bp;
            const bf16x8 hi = *(const bf16x8*)(bp + 16);
#pragma unroll
            for (int i = 0; i < 8; ++i) { bfr[j][i] = lo[i]; bfr[j][i + 8] = hi[i]; }
        }
#pragma unroll
        for (int j = 0; j < 4; ++j)
            acc[j] = __builtin_amdgcn_wmma_f32_16x16x32_bf16(
                         false, afrag, false, bfr[j], (short)0, acc[j], false, false);
    }

    const int m0 = tile_m * 16 + (lane >> 4) * 8;
#pragma unroll
    for (int j = 0; j < 4; ++j) {
        const int tile_n = blockIdx.y * 4 + j;
        if (tile_n < Ntiles) {
            const int nrow = tile_n * 16 + (lane & 15);
            const float bv = bias ? bias[nrow] : 0.0f;
#pragma unroll
            for (int r = 0; r < 8; ++r)
                D[(long long)(m0 + r) * ldd + nrow] = acc[j][r] + bv;
        }
    }
}

// ---------------------------------------------------------------------------
// Additive attention: e[b,l] = sum_a tanh(q[b,a]+k[b,l,a])*v_w[a] + v_b
// then softmax over l. One block (8 waves) per batch row b.
// ---------------------------------------------------------------------------
__global__ void attn_softmax_kernel(const float* __restrict__ q,
                                    const float* __restrict__ kmat,
                                    const float* __restrict__ vw,
                                    const float* __restrict__ vb,
                                    float* __restrict__ attn)
{
    const int b    = blockIdx.x;
    const int tid  = threadIdx.x;
    const int lane = tid & 31;
    const int wave = tid >> 5;
    __shared__ float e[256];
    __shared__ float red[256];

    if (tid >= LL) e[tid] = -3.0e38f;
    const float* qb = q + (size_t)b * AD;
    for (int l = wave; l < LL; l += 8) {
        const float* kr = kmat + ((size_t)b * LL + l) * AD;
        float p = 0.f;
        for (int a = lane; a < AD; a += 32)
            p += tanhf(qb[a] + kr[a]) * vw[a];
#pragma unroll
        for (int off = 16; off; off >>= 1) p += __shfl_xor(p, off, 32);
        if (lane == 0) e[l] = p + vb[0];
    }
    __syncthreads();

    red[tid] = e[tid]; __syncthreads();
    for (int s = 128; s; s >>= 1) { if (tid < s) red[tid] = fmaxf(red[tid], red[tid + s]); __syncthreads(); }
    const float mx = red[0]; __syncthreads();

    const float ex = (tid < LL) ? __expf(e[tid] - mx) : 0.f;
    red[tid] = ex; __syncthreads();
    for (int s = 128; s; s >>= 1) { if (tid < s) red[tid] += red[tid + s]; __syncthreads(); }
    const float inv = 1.0f / red[0];
    if (tid < LL) attn[b * 256 + tid] = ex * inv;
}

// ---------------------------------------------------------------------------
// Build x[b][3072] = [ emb_t (bf16 copy) | ctx = attn·V (fp32->bf16) | (h tail
// written by the LSTM kernel) ].  grid = (10, B), block = 256.
// ---------------------------------------------------------------------------
__global__ void build_x_kernel(const __bf16* __restrict__ emb,
                               const float* __restrict__ attn,
                               const float* __restrict__ V,
                               __bf16* __restrict__ x, int t)
{
    const int b = blockIdx.y;
    const int c = blockIdx.x * 256 + threadIdx.x;
    __shared__ float sa[LL];
    for (int i = threadIdx.x; i < LL; i += 256) sa[i] = attn[b * 256 + i];
    __syncthreads();

    if (c < ED) {
        x[(size_t)b * XK + c] = emb[((size_t)b * NSTEP + t) * ED + c];
    } else if (c < ED + CVD) {
        const int cv = c - ED;
        const float* vp = V + ((size_t)b * LL) * CVD + cv;
        float s = 0.f;
#pragma unroll 4
        for (int l = 0; l < LL; ++l) s += sa[l] * vp[(size_t)l * CVD];
        x[(size_t)b * XK + c] = (__bf16)s;
    }
}

// ---------------------------------------------------------------------------
// LSTM pointwise; writes c (fp32), h (bf16 master) and h into x tail.
// ---------------------------------------------------------------------------
__global__ void lstm_pointwise_kernel(const float* __restrict__ gates,
                                      float* __restrict__ c,
                                      __bf16* __restrict__ hbf,
                                      __bf16* __restrict__ x)
{
    const int idx = blockIdx.x * 256 + threadIdx.x;   // [0, 64*512)
    const int b = idx >> 9, j = idx & 511;
    const float* g = gates + (size_t)b * G4H;
    const float ig = g[j], fg = g[HD + j], gg = g[2 * HD + j], og = g[3 * HD + j];
    const float si = 1.f / (1.f + __expf(-ig));
    const float sf = 1.f / (1.f + __expf(-fg));
    const float so = 1.f / (1.f + __expf(-og));
    const float cn = sf * c[idx] + si * tanhf(gg);
    const float hn = so * tanhf(cn);
    c[idx] = cn;
    const __bf16 hb = (__bf16)hn;
    hbf[idx] = hb;
    x[(size_t)b * XK + (ED + CVD) + j] = hb;
}

// --------------------------- init / utility kernels ------------------------
__global__ void zero_f32_kernel(float* p, int n)
{ int i = blockIdx.x * 256 + threadIdx.x; if (i < n) p[i] = 0.f; }

__global__ void conv_bf16_kernel(const float* __restrict__ s, __bf16* __restrict__ d, int n)
{ int i = blockIdx.x * 256 + threadIdx.x; if (i < n) d[i] = (__bf16)s[i]; }

__global__ void build_wcat_kernel(const float* __restrict__ Wih,
                                  const float* __restrict__ Whh,
                                  __bf16* __restrict__ Wc)
{
    int idx = blockIdx.x * 256 + threadIdx.x;
    if (idx >= G4H * XK) return;
    const int n = idx / XK, cc = idx % XK;
    const float v = (cc < ED + CVD) ? Wih[(size_t)n * (ED + CVD) + cc]
                                    : Whh[(size_t)n * HD + (cc - ED - CVD)];
    Wc[idx] = (__bf16)v;
}

__global__ void bias_sum_kernel(const float* a, const float* b, float* o)
{ int i = blockIdx.x * 256 + threadIdx.x; if (i < G4H) o[i] = a[i] + b[i]; }

__global__ void embed_gather_kernel(const float* __restrict__ EW,
                                    const int* __restrict__ caps,
                                    __bf16* __restrict__ emb)
{
    int idx = blockIdx.x * 256 + threadIdx.x;
    if (idx >= BB * NSTEP * ED) return;
    const int e  = idx & (ED - 1);
    const int bt = idx >> 9;            // b*19 + t
    const int t  = bt % NSTEP, b = bt / NSTEP;
    const int cap = caps[b * TT + t];
    emb[idx] = (__bf16)EW[(size_t)cap * ED + e];
}

// ---------------------------------------------------------------------------
extern "C" void kernel_launch(void* const* d_in, const int* in_sizes, int n_in,
                              void* d_out, int out_size, void* d_ws, size_t ws_size,
                              hipStream_t stream)
{
    (void)in_sizes; (void)n_in; (void)out_size; (void)ws_size;
    const float* V       = (const float*)d_in[0];
    const int*   caps    = (const int*)  d_in[1];
    const float* embed_W = (const float*)d_in[2];
    const float* Wh_w    = (const float*)d_in[3];
    const float* Wh_b    = (const float*)d_in[4];
    const float* Wv_w    = (const float*)d_in[5];
    const float* Wv_b    = (const float*)d_in[6];
    const float* v_w     = (const float*)d_in[7];
    const float* v_b     = (const float*)d_in[8];
    const float* W_ih    = (const float*)d_in[9];
    const float* W_hh    = (const float*)d_in[10];
    const float* b_ih    = (const float*)d_in[11];
    const float* b_hh    = (const float*)d_in[12];
    const float* out_w   = (const float*)d_in[13];
    const float* out_b   = (const float*)d_in[14];
    float* out = (float*)d_out;

    // workspace carve-out (256B aligned slabs), ~54 MB total
    char* ws = (char*)d_ws; size_t off = 0;
    auto carve = [&](size_t bytes) -> void* {
        void* p = ws + off; off = (off + bytes + 255) & ~(size_t)255; return p;
    };
    float*  k_buf   = (float*) carve((size_t)BB * LL * AD * 4);   // attention keys
    __bf16* Wv_bf   = (__bf16*)carve((size_t)AD * CVD * 2);
    __bf16* Wh_bf   = (__bf16*)carve((size_t)AD * HD * 2);
    __bf16* outw_bf = (__bf16*)carve((size_t)VOC * HD * 2);
    __bf16* Wcat_bf = (__bf16*)carve((size_t)G4H * XK * 2);
    float*  bias_g  = (float*) carve((size_t)G4H * 4);
    __bf16* emb_bf  = (__bf16*)carve((size_t)BB * NSTEP * ED * 2);
    __bf16* x_bf    = (__bf16*)carve((size_t)BB * XK * 2);
    __bf16* h_bf    = (__bf16*)carve((size_t)BB * HD * 2);
    float*  c_buf   = (float*) carve((size_t)BB * HD * 4);
    float*  q_buf   = (float*) carve((size_t)BB * AD * 4);
    float*  attn_b  = (float*) carve((size_t)BB * 256 * 4);
    float*  gates_b = (float*) carve((size_t)BB * G4H * 4);

    // ---- one-time setup ----
    conv_bf16_kernel<<<(AD * CVD + 255) / 256, 256, 0, stream>>>(Wv_w, Wv_bf, AD * CVD);
    conv_bf16_kernel<<<(AD * HD + 255) / 256, 256, 0, stream>>>(Wh_w, Wh_bf, AD * HD);
    conv_bf16_kernel<<<(VOC * HD + 255) / 256, 256, 0, stream>>>(out_w, outw_bf, VOC * HD);
    build_wcat_kernel<<<(G4H * XK + 255) / 256, 256, 0, stream>>>(W_ih, W_hh, Wcat_bf);
    bias_sum_kernel<<<(G4H + 255) / 256, 256, 0, stream>>>(b_ih, b_hh, bias_g);
    embed_gather_kernel<<<(BB * NSTEP * ED + 255) / 256, 256, 0, stream>>>(embed_W, caps, emb_bf);
    zero_f32_kernel<<<(BB * HD + 255) / 256, 256, 0, stream>>>(c_buf, BB * HD);
    zero_f32_kernel<<<(BB * HD / 2 + 255) / 256, 256, 0, stream>>>((float*)h_bf, BB * HD / 2);
    zero_f32_kernel<<<(BB * XK / 2 + 255) / 256, 256, 0, stream>>>((float*)x_bf, BB * XK / 2);

    // k = V @ Wv_w^T + Wv_b   [12544 x 512], K=2048, fp32 A converted on the fly
    wmma_gemm_bf16<true><<<dim3(BB * LL / 16 / 4, AD / 16 / 4), 128, 0, stream>>>(
        V, CVD, Wv_bf, CVD, Wv_b, k_buf, (long long)AD, AD / 16, CVD);

    // ---- 19 sequential decode steps ----
    for (int t = 0; t < NSTEP; ++t) {
        // q = h @ Wh_w^T + Wh_b
        wmma_gemm_bf16<false><<<dim3(1, AD / 16 / 4), 128, 0, stream>>>(
            h_bf, HD, Wh_bf, HD, Wh_b, q_buf, (long long)AD, AD / 16, HD);
        // attention + softmax
        attn_softmax_kernel<<<BB, 256, 0, stream>>>(q_buf, k_buf, v_w, v_b, attn_b);
        // x = [emb_t | attn·V | h]
        build_x_kernel<<<dim3((ED + CVD) / 256, BB), 256, 0, stream>>>(emb_bf, attn_b, V, x_bf, t);
        // gates = x @ [W_ih|W_hh]^T + (b_ih+b_hh)
        wmma_gemm_bf16<false><<<dim3(1, G4H / 16 / 4), 128, 0, stream>>>(
            x_bf, XK, Wcat_bf, XK, bias_g, gates_b, (long long)G4H, G4H / 16, XK);
        // LSTM pointwise -> c, h (h also written into x tail for next step)
        lstm_pointwise_kernel<<<(BB * HD) / 256, 256, 0, stream>>>(gates_b, c_buf, h_bf, x_bf);
        // logits_t = h @ out_w^T + out_b   -> d_out[:, t, :]
        wmma_gemm_bf16<false><<<dim3(1, (VOC / 16 + 3) / 4), 128, 0, stream>>>(
            h_bf, HD, outw_bf, HD, out_b, out + (size_t)t * VOC,
            (long long)NSTEP * VOC, VOC / 16, HD);
    }
}
